// MultiHeadAttention_773094113508
// MI455X (gfx1250) — compile-verified, matrix operand order checked by reading, not run on hardware
//
#include <hip/hip_runtime.h>

typedef unsigned short u16;
typedef __attribute__((ext_vector_type(8)))  u16    u16x8;
typedef __attribute__((ext_vector_type(16))) u16    u16x16;
typedef __attribute__((ext_vector_type(16))) __bf16 v16bf;
typedef __attribute__((ext_vector_type(8)))  float  v8f;

#define DM    1024
#define NH    16
#define HD    64
#define SEQ   2048
#define BATCH 2
#define MROWS (BATCH * SEQ) /* 4096 */

// ---------------------------------------------------------------------------
// helpers
// ---------------------------------------------------------------------------
static __device__ __forceinline__ u16 f2bf(float f) {
  unsigned u = __float_as_uint(f);
  u += 0x7FFFu + ((u >> 16) & 1u); // round-to-nearest-even
  return (u16)(u >> 16);
}

// Per-lane WMMA fragment load (A or B side, 16-bit, K=32 chunk).
// Lane L holds row (L&15), k-chunks at +ro and +16+ro where ro=(L>>4)*8.
// Caller passes p = row_ptr + ro; we read p[0..7] and p[16..23].
static __device__ __forceinline__ v16bf frag16(const u16* p) {
  u16x8 lo = *(const u16x8*)(p);
  u16x8 hi = *(const u16x8*)(p + 16);
  u16x16 w = __builtin_shufflevector(lo, hi, 0, 1, 2, 3, 4, 5, 6, 7, 8, 9, 10,
                                     11, 12, 13, 14, 15);
  v16bf r;
  __builtin_memcpy(&r, &w, sizeof(r));
  return r;
}

static __device__ __forceinline__ v8f wmma_bf16(v16bf a, v16bf b, v8f c) {
  // (neg_a, A, neg_b, B, c_mod, C, reuse_a, reuse_b)
  return __builtin_amdgcn_wmma_f32_16x16x32_bf16(false, a, false, b, (short)0,
                                                 c, false, false);
}

static __device__ __forceinline__ v8f vzero8() {
  v8f z;
#pragma unroll
  for (int i = 0; i < 8; ++i) z[i] = 0.0f;
  return z;
}

// reductions across the 16-lane half-groups (wave32: lanes 0-15 / 16-31)
static __device__ __forceinline__ float rmax16(float x) {
  x = fmaxf(x, __shfl_xor(x, 1, 32));
  x = fmaxf(x, __shfl_xor(x, 2, 32));
  x = fmaxf(x, __shfl_xor(x, 4, 32));
  x = fmaxf(x, __shfl_xor(x, 8, 32));
  return x;
}
static __device__ __forceinline__ float rsum16(float x) {
  x += __shfl_xor(x, 1, 32);
  x += __shfl_xor(x, 2, 32);
  x += __shfl_xor(x, 4, 32);
  x += __shfl_xor(x, 8, 32);
  return x;
}

// ---------------------------------------------------------------------------
// fp32 -> bf16 conversion
// ---------------------------------------------------------------------------
__global__ void f2bf_kernel(const float* __restrict__ in, u16* __restrict__ out,
                            int n) {
  int i = blockIdx.x * blockDim.x + threadIdx.x;
  int stride = gridDim.x * blockDim.x;
  for (; i < n; i += stride) out[i] = f2bf(in[i]);
}

// ---------------------------------------------------------------------------
// GEMM: out = A[M,K] @ W[N,K]^T + bias, bf16 inputs, fp32 accumulate.
// M=4096, N=K=1024 hardcoded. 256 threads = 8 waves; block tile 128x128;
// each wave computes a 32x64 sub-tile (2x4 WMMA tiles).
// mode 0: fp32 out [M,N]           (final projection -> d_out)
// mode 1: bf16 out [B,H,S,hd]      (Q, K head-split)
// mode 2: bf16 out [B,H,hd,S]      (V head-split, pre-transposed for PV WMMA)
// ---------------------------------------------------------------------------
__global__ __launch_bounds__(256) void gemm_bf16_kernel(
    const u16* __restrict__ A, const u16* __restrict__ W,
    const float* __restrict__ bias, u16* __restrict__ outb,
    float* __restrict__ outf, int mode) {
  const int lane = threadIdx.x & 31;
  const int wave = threadIdx.x >> 5;
  const int r  = lane & 15;         // fragment row / C column index
  const int ro = (lane >> 4) * 8;   // k-chunk offset == C row base
  const int m0 = blockIdx.y * 128 + (wave >> 1) * 32;
  const int n0 = blockIdx.x * 128 + (wave & 1) * 64;

  v8f acc[2][4];
#pragma unroll
  for (int mt = 0; mt < 2; ++mt)
#pragma unroll
    for (int nt = 0; nt < 4; ++nt) acc[mt][nt] = vzero8();

  const u16* arow0 = A + (size_t)(m0 + r) * DM;
  const u16* arow1 = A + (size_t)(m0 + 16 + r) * DM;
  const u16* wrow[4];
#pragma unroll
  for (int nt = 0; nt < 4; ++nt)
    wrow[nt] = W + (size_t)(n0 + nt * 16 + r) * DM;

  for (int kb = 0; kb < DM; kb += 32) {
    v16bf a0 = frag16(arow0 + kb + ro);
    v16bf a1 = frag16(arow1 + kb + ro);
#pragma unroll
    for (int nt = 0; nt < 4; ++nt) {
      v16bf bf = frag16(wrow[nt] + kb + ro);
      acc[0][nt] = wmma_bf16(a0, bf, acc[0][nt]);
      acc[1][nt] = wmma_bf16(a1, bf, acc[1][nt]);
    }
  }

#pragma unroll
  for (int mt = 0; mt < 2; ++mt) {
#pragma unroll
    for (int nt = 0; nt < 4; ++nt) {
      const int n = n0 + nt * 16 + r; // lane's C column
      const float bn = bias[n];
#pragma unroll
      for (int v = 0; v < 8; ++v) {
        const int m = m0 + mt * 16 + ro + v; // C row
        const float val = acc[mt][nt][v] + bn;
        if (mode == 0) {
          outf[(size_t)m * DM + n] = val;
        } else {
          const int b = m >> 11, s = m & (SEQ - 1);
          const int h = n >> 6, d = n & (HD - 1);
          size_t idx =
              (mode == 1)
                  ? ((size_t)(b * NH + h) * SEQ + s) * HD + d   // [B,H,S,hd]
                  : ((size_t)(b * NH + h) * HD + d) * SEQ + s;  // [B,H,hd,S]
          outb[idx] = f2bf(val);
        }
      }
    }
  }
}

// ---------------------------------------------------------------------------
// Causal flash attention. One wave per 16-query tile; 4 waves/block, each
// independent (private LDS region, no cross-wave sync). Online softmax over
// 32-key steps: 4 score WMMAs + LDS P-transpose + 4 PV WMMAs per step.
// Q,K: bf16 [B,H,S,hd]; Vt: bf16 [B,H,hd,S]; O: bf16 [B,S,D] (heads merged).
// ---------------------------------------------------------------------------
__global__ __launch_bounds__(128) void flash_attn_kernel(
    const u16* __restrict__ Q, const u16* __restrict__ K,
    const u16* __restrict__ Vt, u16* __restrict__ O) {
  __shared__ u16 smem[4 * 16 * 32]; // per-wave 16x32 bf16 P tile
  const int lane = threadIdx.x & 31;
  const int w = threadIdx.x >> 5;
  const int r  = lane & 15;
  const int ro = (lane >> 4) * 8;
  const int b = blockIdx.z, h = blockIdx.y;
  const int q0 = (blockIdx.x * 4 + w) * 16;

  const u16* Qb = Q + (size_t)(b * NH + h) * SEQ * HD;
  const u16* Kb = K + (size_t)(b * NH + h) * SEQ * HD;
  const u16* Vb = Vt + (size_t)(b * NH + h) * HD * SEQ;
  u16* lp = &smem[w * (16 * 32)];

  // Q fragments (reused for every key step)
  const v16bf aql = frag16(Qb + (size_t)(q0 + r) * HD + 0 + ro);
  const v16bf aqh = frag16(Qb + (size_t)(q0 + r) * HD + 32 + ro);

  v8f o[4];
#pragma unroll
  for (int dt = 0; dt < 4; ++dt) o[dt] = vzero8();
  float m_r[8], l_r[8];
#pragma unroll
  for (int v = 0; v < 8; ++v) { m_r[v] = -1e30f; l_r[v] = 0.0f; }

  const float scale = 0.125f; // 1/sqrt(64)

  for (int j = 0; j <= q0 + 15; j += 32) {
    const u16* k0row = Kb + (size_t)(j + r) * HD;
    const u16* k1row = Kb + (size_t)(j + 16 + r) * HD;

    // scores for key tiles [j, j+16) and [j+16, j+32), K-dim 64 = 2 WMMAs
    v8f s0 = wmma_bf16(aqh, frag16(k0row + 32 + ro), vzero8());
    s0 = wmma_bf16(aql, frag16(k0row + 0 + ro), s0);
    v8f s1 = wmma_bf16(aqh, frag16(k1row + 32 + ro), vzero8());
    s1 = wmma_bf16(aql, frag16(k1row + 0 + ro), s1);

    float corr[8];
#pragma unroll
    for (int v = 0; v < 8; ++v) {
      const int qi = q0 + ro + v; // this lane-half's row = query index
      float x0 = s0[v] * scale;
      if (j + r > qi) x0 = -1e30f; // causal mask, key index = j + r
      float x1 = s1[v] * scale;
      if (j + 16 + r > qi) x1 = -1e30f;
      const float mrow = rmax16(fmaxf(x0, x1));
      const float mnew = fmaxf(m_r[v], mrow);
      const float c = __expf(m_r[v] - mnew);
      const float p0 = __expf(x0 - mnew);
      const float p1 = __expf(x1 - mnew);
      l_r[v] = l_r[v] * c + rsum16(p0 + p1);
      m_r[v] = mnew;
      corr[v] = c;
      // scatter P (C layout) into row-major LDS tile for A-fragment reload
      lp[(ro + v) * 32 + r] = f2bf(p0);
      lp[(ro + v) * 32 + 16 + r] = f2bf(p1);
    }

    // rescale running output by softmax correction (overlaps with DS stores)
#pragma unroll
    for (int dt = 0; dt < 4; ++dt)
#pragma unroll
      for (int v = 0; v < 8; ++v) o[dt][v] *= corr[v];

    // same-wave DS ordering: drain LDS stores before fragment reload
    asm volatile("s_wait_dscnt 0x0" ::: "memory");
    const v16bf ap = frag16(lp + r * 32 + ro); // P as 16x32 A fragment
#pragma unroll
    for (int dt = 0; dt < 4; ++dt) {
      v16bf bv = frag16(Vb + (size_t)(dt * 16 + r) * SEQ + j + ro);
      o[dt] = wmma_bf16(ap, bv, o[dt]);
    }
  }

  float rl[8];
#pragma unroll
  for (int v = 0; v < 8; ++v) rl[v] = 1.0f / l_r[v];

#pragma unroll
  for (int dt = 0; dt < 4; ++dt) {
    const int col = h * HD + dt * 16 + r;
#pragma unroll
    for (int v = 0; v < 8; ++v) {
      const int s = q0 + ro + v;
      O[((size_t)(b * SEQ + s)) * DM + col] = f2bf(o[dt][v] * rl[v]);
    }
  }
}

// ---------------------------------------------------------------------------
// launch
// ---------------------------------------------------------------------------
extern "C" void kernel_launch(void* const* d_in, const int* in_sizes, int n_in,
                              void* d_out, int out_size, void* d_ws,
                              size_t ws_size, hipStream_t stream) {
  (void)in_sizes; (void)n_in; (void)out_size; (void)ws_size;
  const float* q  = (const float*)d_in[0];
  const float* k  = (const float*)d_in[1];
  const float* v  = (const float*)d_in[2];
  const float* Wq = (const float*)d_in[3];
  const float* bq = (const float*)d_in[4];
  const float* Wk = (const float*)d_in[5];
  const float* bk = (const float*)d_in[6];
  const float* Wv = (const float*)d_in[7];
  const float* bv = (const float*)d_in[8];
  const float* Wo = (const float*)d_in[9];
  const float* bo = (const float*)d_in[10];

  const size_t NACT = (size_t)MROWS * DM; // 4 Mi elements
  const size_t NW   = (size_t)DM * DM;    // 1 Mi elements

  u16* qbf = (u16*)d_ws;
  u16* kbf = qbf + NACT;
  u16* vbf = kbf + NACT;
  u16* wqb = vbf + NACT;
  u16* wkb = wqb + NW;
  u16* wvb = wkb + NW;
  u16* wob = wvb + NW;
  u16* Qh  = wob + NW;   // [B,H,S,hd] bf16
  u16* Kh  = Qh + NACT;  // [B,H,S,hd] bf16
  u16* Vt  = Kh + NACT;  // [B,H,hd,S] bf16
  u16* Ob  = Vt + NACT;  // [B,S,D]    bf16 (attention output, heads merged)

  f2bf_kernel<<<2048, 256, 0, stream>>>(q, qbf, (int)NACT);
  f2bf_kernel<<<2048, 256, 0, stream>>>(k, kbf, (int)NACT);
  f2bf_kernel<<<2048, 256, 0, stream>>>(v, vbf, (int)NACT);
  f2bf_kernel<<<1024, 256, 0, stream>>>(Wq, wqb, (int)NW);
  f2bf_kernel<<<1024, 256, 0, stream>>>(Wk, wkb, (int)NW);
  f2bf_kernel<<<1024, 256, 0, stream>>>(Wv, wvb, (int)NW);
  f2bf_kernel<<<1024, 256, 0, stream>>>(Wo, wob, (int)NW);

  dim3 gg(DM / 128, MROWS / 128); // (8, 32)
  gemm_bf16_kernel<<<gg, 256, 0, stream>>>(qbf, wqb, bq, Qh, nullptr, 1);
  gemm_bf16_kernel<<<gg, 256, 0, stream>>>(kbf, wkb, bk, Kh, nullptr, 1);
  gemm_bf16_kernel<<<gg, 256, 0, stream>>>(vbf, wvb, bv, Vt, nullptr, 2);

  flash_attn_kernel<<<dim3(SEQ / 64, NH, BATCH), 128, 0, stream>>>(Qh, Kh, Vt,
                                                                   Ob);

  gemm_bf16_kernel<<<gg, 256, 0, stream>>>(Ob, wob, bo, nullptr,
                                           (float*)d_out, 0);
}
